// MultiLayerGCN_11312943858128
// MI455X (gfx1250) — matrix operand order repaired and checked by reading, not compiled, not run on hardware
//
#include <hip/hip_runtime.h>

#define NN 50000
#define NE 800000
#define D_IN 128
#define D_HID 128
#define D_OUT 64

typedef __attribute__((ext_vector_type(16))) __bf16 v16bf;
typedef __attribute__((ext_vector_type(8)))  __bf16 v8bf;
typedef __attribute__((ext_vector_type(8)))  float  v8f;

// ---------------- small utility kernels ----------------

__global__ void cvt_f32_to_bf16(const float* __restrict__ src, __bf16* __restrict__ dst, int n) {
  int i = blockIdx.x * blockDim.x + threadIdx.x;
  if (i < n) dst[i] = (__bf16)src[i];
}

__global__ void fill_f32(float* __restrict__ p, float v, int n) {
  int i = blockIdx.x * blockDim.x + threadIdx.x;
  if (i < n) p[i] = v;
}

__global__ void deg_accum(const long long* __restrict__ dst_idx, float* __restrict__ deg, int e) {
  int i = blockIdx.x * blockDim.x + threadIdx.x;
  if (i < e) unsafeAtomicAdd(&deg[(int)dst_idx[i]], 1.0f);
}

__global__ void rsqrt_inplace(float* __restrict__ p, int n) {
  int i = blockIdx.x * blockDim.x + threadIdx.x;
  if (i < n) p[i] = rsqrtf(p[i]);
}

// ---------------- WMMA GEMM: C[M,N] = A[M,128](bf16) x B[128,N](bf16), f32 accum ----------------
// Block = 256 threads = 8 waves; one 16-col N tile per block (blockIdx.y).
// Stage the 128x16 bf16 W tile (4 KB) into LDS in *fragment order*:
//   element (k, n_local) -> smem[(chunk*32 + L)*16 + e], chunk=k/32, L=n_local+16*((k>>4)&1), e=k&15
// so a lane's v16bf B fragment for chunk c is one contiguous 32B LDS read (2x ds_load_b128).
// Each wave then sweeps 4 consecutive 16-row M tiles reusing the LDS-resident B fragments.
//
// A fragment (16-bit A 16x32): lane L -> row m = L%16, half = L/16;
//   elems 0..7  -> k = c*32 + half*16 + e          (one contiguous 16B load)
//   elems 8..15 -> k = c*32 + half*16 + 8 + (e-8) (one contiguous 16B load)
// C tile: VGPR r, lane L -> row (L/16)*8 + r, col L%16.
template <int N>
__global__ void gemm_bf16_wmma(const __bf16* __restrict__ A, const __bf16* __restrict__ B,
                               float* __restrict__ C, int M) {
  const int K = 128;
  __shared__ __align__(32) __bf16 smem[4 * 32 * 16];  // 4 KB: [chunk][lane][elem]

  int tid  = threadIdx.x;
  int lane = tid & 31;
  int wave = tid >> 5;  // 0..7
  int nbase = blockIdx.y * 16;
  int half = lane >> 4;
  int l16  = lane & 15;

  // ---- cooperative staging: one coalesced b128 load + 8 ds_store_b16 per thread ----
  {
    int k  = tid >> 1;         // 0..127 (row of W)
    int h8 = (tid & 1) << 3;   // n_local 0 or 8
    v8bf w = *(const v8bf*)(B + (size_t)k * N + nbase + h8);
    int c = k >> 5, hb = (k >> 4) & 1, e = k & 15;
#pragma unroll
    for (int j = 0; j < 8; ++j) {
      int L = h8 + j + (hb << 4);
      smem[((c * 32 + L) << 4) + e] = w[j];
    }
  }
  __syncthreads();

  // ---- B fragments: contiguous 32B LDS reads, kept across the M sweep ----
  v16bf bf[4];
#pragma unroll
  for (int c = 0; c < 4; ++c) bf[c] = *(const v16bf*)(smem + ((c * 32 + lane) << 4));

  int mtile0 = (blockIdx.x * 8 + wave) * 4;  // first 16-row tile for this wave
#pragma unroll
  for (int t = 0; t < 4; ++t) {
    int mbase = (mtile0 + t) * 16;
    if (mbase >= M) break;  // wave-uniform: EXEC stays all-ones
    const __bf16* __restrict__ arow = A + (size_t)(mbase + l16) * K + half * 16;
    v8f acc = {};
#pragma unroll
    for (int c = 0; c < 4; ++c) {
      v8bf lo = *(const v8bf*)(arow + c * 32);
      v8bf hi = *(const v8bf*)(arow + c * 32 + 8);
      v16bf a = __builtin_shufflevector(lo, hi, 0,1,2,3,4,5,6,7,8,9,10,11,12,13,14,15);
      acc = __builtin_amdgcn_wmma_f32_16x16x32_bf16(false, a, false, bf[c], (short)0, acc,
                                                    false, false);
    }
    float* __restrict__ crow = C + (size_t)(mbase + half * 8) * N + nbase + l16;
#pragma unroll
    for (int r = 0; r < 8; ++r) crow[r * N] = acc[r];
  }
}

// ---------------- edge scatter: agg[dst] += h[src] * dinv[src]*dinv[dst] (A + I) ----------------
__global__ void scatter_edges(const float* __restrict__ h,
                              const long long* __restrict__ src_idx,
                              const long long* __restrict__ dst_idx,
                              const float* __restrict__ dinv,
                              float* __restrict__ agg,
                              int n_edges, int n_nodes, int dim) {
  int chunks = dim >> 2;  // 4 floats per thread
  long long tid = (long long)blockIdx.x * blockDim.x + threadIdx.x;
  long long total = (long long)(n_edges + n_nodes) * chunks;
  if (tid >= total) return;
  int e = (int)(tid / chunks);
  int f = ((int)(tid % chunks)) << 2;
  int s, d;
  if (e < n_edges) { s = (int)src_idx[e]; d = (int)dst_idx[e]; }
  else             { s = d = e - n_edges; }  // self loop
  float norm = dinv[s] * dinv[d];
  const float4 hv = *(const float4*)(h + (size_t)s * dim + f);
  float* ap = agg + (size_t)d * dim + f;
  unsafeAtomicAdd(ap + 0, hv.x * norm);
  unsafeAtomicAdd(ap + 1, hv.y * norm);
  unsafeAtomicAdd(ap + 2, hv.z * norm);
  unsafeAtomicAdd(ap + 3, hv.w * norm);
}

// ---------------- finalize ----------------

__global__ void bias_relu_to_bf16(const float* __restrict__ agg, const float* __restrict__ bias,
                                  __bf16* __restrict__ out, int total, int dimMask) {
  int i = blockIdx.x * blockDim.x + threadIdx.x;
  if (i >= total) return;
  float v = agg[i] + bias[i & dimMask];
  out[i] = (__bf16)fmaxf(v, 0.0f);
}

__global__ void bias_to_f32(const float* __restrict__ agg, const float* __restrict__ bias,
                            float* __restrict__ out, int total, int dimMask) {
  int i = blockIdx.x * blockDim.x + threadIdx.x;
  if (i >= total) return;
  out[i] = agg[i] + bias[i & dimMask];
}

// ---------------- host ----------------

static inline unsigned nblk(long long n, int b) { return (unsigned)((n + b - 1) / b); }

extern "C" void kernel_launch(void* const* d_in, const int* in_sizes, int n_in,
                              void* d_out, int out_size, void* d_ws, size_t ws_size,
                              hipStream_t stream) {
  const float*     x  = (const float*)d_in[0];
  const long long* ei = (const long long*)d_in[1];
  const float*     W1 = (const float*)d_in[2];
  const float*     b1 = (const float*)d_in[3];
  const float*     W2 = (const float*)d_in[4];
  const float*     b2 = (const float*)d_in[5];
  const float*     W3 = (const float*)d_in[6];
  const float*     b3 = (const float*)d_in[7];
  float* out = (float*)d_out;

  char* ws = (char*)d_ws;
  float*  dinv = (float*)(ws + 0);                 // 50000 f32
  __bf16* Wb   = (__bf16*)(ws + (size_t)262144);   // 40960 bf16 (W1b,W2b,W3b)
  __bf16* xb   = (__bf16*)(ws + (size_t)524288);   // 50000*128 bf16 (12.8 MB)
  float*  h    = (float*)(ws + (size_t)14680064);  // 50000*128 f32 (25.6 MB)
  float*  agg  = (float*)(ws + (size_t)41943040);  // 50000*128 f32 (25.6 MB)
  __bf16* W1b = Wb;
  __bf16* W2b = Wb + D_IN * D_HID;
  __bf16* W3b = Wb + D_IN * D_HID + D_HID * D_HID;

  const long long* srcI = ei;       // edge_index[0]
  const long long* dstI = ei + NE;  // edge_index[1]

  const int T = 256;

  // one-time conversions (per launch; deterministic)
  cvt_f32_to_bf16<<<nblk(D_IN * D_HID, T), T, 0, stream>>>(W1, W1b, D_IN * D_HID);
  cvt_f32_to_bf16<<<nblk(D_HID * D_HID, T), T, 0, stream>>>(W2, W2b, D_HID * D_HID);
  cvt_f32_to_bf16<<<nblk(D_HID * D_OUT, T), T, 0, stream>>>(W3, W3b, D_HID * D_OUT);
  cvt_f32_to_bf16<<<nblk((long long)NN * D_IN, T), T, 0, stream>>>(x, xb, NN * D_IN);

  // degree -> dinv = rsqrt(deg), deg includes self loop
  fill_f32<<<nblk(NN, T), T, 0, stream>>>(dinv, 1.0f, NN);
  deg_accum<<<nblk(NE, T), T, 0, stream>>>(dstI, dinv, NE);
  rsqrt_inplace<<<nblk(NN, T), T, 0, stream>>>(dinv, NN);

  // ---- layer 1: h = xb @ W1b ; agg = scatter ; xb = bf16(relu(agg + b1)) ----
  {
    dim3 g(nblk(NN, 512), D_HID / 16);
    gemm_bf16_wmma<D_HID><<<g, 256, 0, stream>>>(xb, W1b, h, NN);
    fill_f32<<<nblk((long long)NN * D_HID, T), T, 0, stream>>>(agg, 0.0f, NN * D_HID);
    scatter_edges<<<nblk((long long)(NE + NN) * (D_HID / 4), T), T, 0, stream>>>(
        h, srcI, dstI, dinv, agg, NE, NN, D_HID);
    bias_relu_to_bf16<<<nblk((long long)NN * D_HID, T), T, 0, stream>>>(
        agg, b1, xb, NN * D_HID, D_HID - 1);
  }

  // ---- layer 2 ----
  {
    dim3 g(nblk(NN, 512), D_HID / 16);
    gemm_bf16_wmma<D_HID><<<g, 256, 0, stream>>>(xb, W2b, h, NN);
    fill_f32<<<nblk((long long)NN * D_HID, T), T, 0, stream>>>(agg, 0.0f, NN * D_HID);
    scatter_edges<<<nblk((long long)(NE + NN) * (D_HID / 4), T), T, 0, stream>>>(
        h, srcI, dstI, dinv, agg, NE, NN, D_HID);
    bias_relu_to_bf16<<<nblk((long long)NN * D_HID, T), T, 0, stream>>>(
        agg, b2, xb, NN * D_HID, D_HID - 1);
  }

  // ---- layer 3 (no relu, f32 output) ----
  {
    dim3 g(nblk(NN, 512), D_OUT / 16);
    gemm_bf16_wmma<D_OUT><<<g, 256, 0, stream>>>(xb, W3b, h, NN);
    fill_f32<<<nblk((long long)NN * D_OUT, T), T, 0, stream>>>(agg, 0.0f, NN * D_OUT);
    scatter_edges<<<nblk((long long)(NE + NN) * (D_OUT / 4), T), T, 0, stream>>>(
        h, srcI, dstI, dinv, agg, NE, NN, D_OUT);
    bias_to_f32<<<nblk((long long)NN * D_OUT, T), T, 0, stream>>>(
        agg, b3, out, NN * D_OUT, D_OUT - 1);
  }
}
